// RPN_79929341378759
// MI455X (gfx1250) — matrix operand order repaired and checked by reading, not compiled, not run on hardware
//
#include <hip/hip_runtime.h>

typedef __attribute__((ext_vector_type(16))) __bf16 v16bf;
typedef __attribute__((ext_vector_type(8)))  float  v8f;
typedef unsigned int v4u __attribute__((ext_vector_type(4)));

#define NEGINF (-__builtin_inff())

// fp32 -> bf16, round-to-nearest-even
static __device__ inline unsigned f2bf(float f) {
    unsigned u = __float_as_uint(f);
    return (u + 0x7FFFu + ((u >> 16) & 1u)) >> 16;
}

// ---------------- prep: pad & convert x (NCHW f32) -> x_pad (N,66,66,512) bf16 ----------------
__global__ void k_pad_x(const float* __restrict__ x, unsigned short* __restrict__ xpad) {
    long gid = (long)blockIdx.x * blockDim.x + threadIdx.x;
    if (gid >= 17842176L) return;
    int c   = gid & 511;
    long r  = gid >> 9;
    int rx  = r % 66; r /= 66;
    int ry  = r % 66;
    int n   = r / 66;
    unsigned v = 0;
    if (ry >= 1 && ry <= 64 && rx >= 1 && rx <= 64) {
        long src = (((long)n * 512 + c) * 64 + (ry - 1)) * 64 + (rx - 1);
        v = f2bf(x[src]);
    }
    xpad[gid] = (unsigned short)v;
}

// ---------------- prep: pack conv1 weights into A-fragment layout ----------------
// frag id = (t*16 + kc)*32 + mt ; within frag: dword index = lane*8 + r
__global__ void k_pack_w1(const float* __restrict__ w1, unsigned* __restrict__ wp) {
    long gid = (long)blockIdx.x * blockDim.x + threadIdx.x;
    if (gid >= 1179648L) return;
    int r    = gid & 7;
    int lane = (gid >> 3) & 31;
    int frag = gid >> 8;
    int t    = frag / 512;
    int rem  = frag - t * 512;
    int kc   = rem >> 5;
    int mt   = rem & 31;
    int h    = lane >> 4;
    int m    = mt * 16 + (lane & 15);
    int kb   = 16 * h + 2 * r;
    int c    = kc * 32 + kb;
    unsigned lo = f2bf(w1[((long)m * 512 + c) * 9 + t]);
    unsigned hi = f2bf(w1[((long)m * 512 + (c + 1)) * 9 + t]);
    wp[gid] = lo | (hi << 16);
}

// ---------------- prep: pack 1x1 head weights (M padded to mtiles*16) ----------------
__global__ void k_pack_head(const float* __restrict__ w, unsigned* __restrict__ wp, int M, int mtiles) {
    long gid = (long)blockIdx.x * blockDim.x + threadIdx.x;
    long tot = (long)16 * mtiles * 256;
    if (gid >= tot) return;
    int r    = gid & 7;
    int lane = (gid >> 3) & 31;
    int frag = gid >> 8;
    int kc   = frag / mtiles;
    int mt   = frag - kc * mtiles;
    int h    = lane >> 4;
    int m    = mt * 16 + (lane & 15);
    int kb   = 16 * h + 2 * r;
    int c    = kc * 32 + kb;
    unsigned lo = 0, hi = 0;
    if (m < M) {
        lo = f2bf(w[(long)m * 512 + c]);
        hi = f2bf(w[(long)m * 512 + c + 1]);
    }
    wp[gid] = lo | (hi << 16);
}

// ---------------- conv1: 3x3, 512->512, relu; implicit GEMM, 2x2 register-blocked WMMA ----------
// Block = 8 waves = 2 Mw x 4 Nw; wave computes 32M x 32N (4 accumulators).
// Grid: x = 32 (128 positions each), y = 8 (64 channels each), z = 8 images.
__global__ __launch_bounds__(256) void k_conv1(const unsigned short* __restrict__ xpad,
                                               const v4u* __restrict__ wp,
                                               const float* __restrict__ b1,
                                               unsigned short* __restrict__ feat) {
    const int lane = threadIdx.x & 31;
    const int wave = threadIdx.x >> 5;
    const int nimg = blockIdx.z;
    const int mw   = wave >> 2;           // 0..1
    const int nw   = wave & 3;            // 0..3
    const int mtA  = blockIdx.y * 4 + mw * 2;   // 0..30 (even)
    const int ntA  = blockIdx.x * 8 + nw * 2;   // 0..254 (even)
    const int nl   = lane & 15, hh = lane >> 4;
    const int posb = ntA * 16;                  // multiple of 32 -> both tiles in one row
    const int y    = posb >> 6, xb = posb & 63;

    v8f acc00 = {}, acc01 = {}, acc10 = {}, acc11 = {};
    const size_t imgbase = (size_t)nimg * 4356 * 512;

    for (int t = 0; t < 9; ++t) {
        const int dy = t / 3, dx = t - dy * 3;
        // B base: element (row, xb+dx+nl, channel hh*16); second N-tile at +16 cols = +16384 B
        const v4u* bp = (const v4u*)(xpad + imgbase
                                     + (size_t)((y + dy) * 66 + xb + dx + nl) * 512 + hh * 16);
        // A base: frag (t*16 + 0)*32 + mtA; per-kc stride = 32 frags = 2048 v4u
        const v4u* ap = wp + (size_t)(t * 16 * 32 + mtA) * 64 + lane * 2;
#pragma unroll 2
        for (int kc = 0; kc < 16; ++kc) {
            union { v4u q[2]; v16bf v; } A0, A1, B0, B1;
            A0.q[0] = ap[0];    A0.q[1] = ap[1];       // M-tile mtA
            A1.q[0] = ap[64];   A1.q[1] = ap[65];      // M-tile mtA+1 (+1024 B)
            B0.q[0] = bp[0];    B0.q[1] = bp[1];       // N-tile ntA
            B1.q[0] = bp[1024]; B1.q[1] = bp[1025];    // N-tile ntA+1 (+16384 B)
            acc00 = __builtin_amdgcn_wmma_f32_16x16x32_bf16(false, A0.v, false, B0.v,
                                                            (short)0, acc00, false, false);
            acc01 = __builtin_amdgcn_wmma_f32_16x16x32_bf16(false, A0.v, false, B1.v,
                                                            (short)0, acc01, false, false);
            acc10 = __builtin_amdgcn_wmma_f32_16x16x32_bf16(false, A1.v, false, B0.v,
                                                            (short)0, acc10, false, false);
            acc11 = __builtin_amdgcn_wmma_f32_16x16x32_bf16(false, A1.v, false, B1.v,
                                                            (short)0, acc11, false, false);
            ap += 2048;   // next 32-channel chunk of weights (32 frags * 1024 B)
            bp += 4;      // next 32 channels of x (64 B)
        }
    }

    // epilogue: bias + relu; D layout m = v + 8*hh -> 8 consecutive channels per lane
    const int pos0 = posb + nl;
    const float* bb = b1;
#pragma unroll
    for (int mi = 0; mi < 2; ++mi) {
        const int mstart = (mtA + mi) * 16 + 8 * hh;
        v8f accA = mi ? acc10 : acc00;   // N-tile ntA
        v8f accB = mi ? acc11 : acc01;   // N-tile ntA+1
        unsigned wa[4], wb[4];
#pragma unroll
        for (int p = 0; p < 4; ++p) {
            float bias0 = bb[mstart + 2 * p], bias1 = bb[mstart + 2 * p + 1];
            float f0 = accA[2 * p] + bias0;     f0 = f0 > 0.f ? f0 : 0.f;
            float f1 = accA[2 * p + 1] + bias1; f1 = f1 > 0.f ? f1 : 0.f;
            wa[p] = f2bf(f0) | (f2bf(f1) << 16);
            float g0 = accB[2 * p] + bias0;     g0 = g0 > 0.f ? g0 : 0.f;
            float g1 = accB[2 * p + 1] + bias1; g1 = g1 > 0.f ? g1 : 0.f;
            wb[p] = f2bf(g0) | (f2bf(g1) << 16);
        }
        v4u oa = {wa[0], wa[1], wa[2], wa[3]};
        v4u ob = {wb[0], wb[1], wb[2], wb[3]};
        *(v4u*)(feat + (size_t)(nimg * 4096 + pos0) * 512 + mstart) = oa;
        *(v4u*)(feat + (size_t)(nimg * 4096 + pos0 + 16) * 512 + mstart) = ob;
    }
}

// ---------------- heads (1x1 convs via WMMA) + proposal decode ----------------
__global__ __launch_bounds__(256) void k_head(const unsigned short* __restrict__ feat,
                                              const v4u* __restrict__ wsp,
                                              const v4u* __restrict__ wlp,
                                              const float* __restrict__ bscore,
                                              const float* __restrict__ bloc,
                                              const int* __restrict__ imh,
                                              const int* __restrict__ imw,
                                              float* __restrict__ out_locs,
                                              float* __restrict__ out_scores,
                                              float* __restrict__ out_anchor,
                                              float* __restrict__ boxes,
                                              float* __restrict__ scinit) {
    __shared__ float sraw[32][16];
    __shared__ float lraw[48][16];
    const int lane = threadIdx.x & 31;
    const int wave = threadIdx.x >> 5;
    const int nimg = blockIdx.y;
    const int posb = blockIdx.x * 16;
    const int nl   = lane & 15, hh = lane >> 4;

    if (wave < 5) {
        const bool isScore = wave < 2;
        const int mt      = isScore ? wave : wave - 2;
        const int mtiles  = isScore ? 2 : 3;
        const v4u* wp     = isScore ? wsp : wlp;
        v8f acc = {};
        const size_t fbase = (size_t)(nimg * 4096 + posb + nl) * 512;
#pragma unroll 4
        for (int kc = 0; kc < 16; ++kc) {
            union { v4u q[2]; v16bf v; } A, B;
            const v4u* ap = wp + (size_t)(kc * mtiles + mt) * 64 + lane * 2;
            A.q[0] = ap[0]; A.q[1] = ap[1];
            const v4u* bp = (const v4u*)(feat + fbase + kc * 32 + hh * 16);
            B.q[0] = bp[0]; B.q[1] = bp[1];
            acc = __builtin_amdgcn_wmma_f32_16x16x32_bf16(false, A.v, false, B.v,
                                                          (short)0, acc, false, false);
        }
#pragma unroll
        for (int v = 0; v < 8; ++v) {
            int m = mt * 16 + v + 8 * hh;
            if (isScore) { if (m < 18) sraw[m][nl] = acc[v] + bscore[m]; }
            else         { if (m < 36) lraw[m][nl] = acc[v] + bloc[m];   }
        }
    }
    __syncthreads();

    const int tid = threadIdx.x;
    if (tid < 144) {
        const int p = tid / 9, t = tid - 9 * p;
        const int pos = posb + p;
        const long a  = (long)pos * 9 + t;
        const float fW = (float)imw[0], fH = (float)imh[0];
        const float tx = lraw[4 * t][p],     ty = lraw[4 * t + 1][p];
        const float tw = lraw[4 * t + 2][p], th = lraw[4 * t + 3][p];
        const float s0 = sraw[2 * t][p],     s1 = sraw[2 * t + 1][p];
        const long lb = (long)nimg * 36864 + a;
        out_locs[lb * 4 + 0] = tx; out_locs[lb * 4 + 1] = ty;
        out_locs[lb * 4 + 2] = tw; out_locs[lb * 4 + 3] = th;
        out_scores[lb * 2 + 0] = s0; out_scores[lb * 2 + 1] = s1;
        // anchor base: ratios {0.5,1,2} x scales {8,16,32}; h=s*sqrt(r), w=s*sqrt(1/r)
        const float sqr[3]  = {0.70710678118654752f, 1.0f, 1.41421356237309505f};
        const float sqir[3] = {1.41421356237309505f, 1.0f, 0.70710678118654752f};
        const int i = t / 3, j = t - 3 * i;
        const float sc = (float)(8 << j);
        const float ah = sc * sqr[i], aw = sc * sqir[i];
        const float sx = (float)((pos & 63) * 16), sy = (float)((pos >> 6) * 16);
        const float a0 = sx - 0.5f * ah, a1 = sy - 0.5f * aw;
        const float a2 = sx + 0.5f * ah, a3 = sy + 0.5f * aw;
        const float px = tx * 16.f + a0, py = ty * 16.f + a1;
        const float pw = expf(tw) * ah,  ph = expf(th) * aw;
        const float x1 = fminf(fmaxf(px, 0.f), fW);
        const float x2 = fminf(fmaxf(px + pw, 0.f), fW);
        const float y1 = fminf(fmaxf(py, 0.f), fH);
        const float y2 = fminf(fmaxf(py + ph, 0.f), fH);
        const bool valid = (x2 - x1 + 1.f >= 16.f) && (y2 - y1 + 1.f >= 16.f);
        const float fg = 1.f / (1.f + expf(s0 - s1));   // softmax class-1 prob
        boxes[lb * 4 + 0] = x1; boxes[lb * 4 + 1] = y1;
        boxes[lb * 4 + 2] = x2; boxes[lb * 4 + 3] = y2;
        scinit[lb] = valid ? fg : NEGINF;
        if (nimg == 0) {
            out_anchor[a * 4 + 0] = a0; out_anchor[a * 4 + 1] = a1;
            out_anchor[a * 4 + 2] = a2; out_anchor[a * 4 + 3] = a3;
        }
    }
}

// ---------------- NMS: one block per image, reference-exact serial loop ----------------
__global__ __launch_bounds__(1024) void k_nms(const float* __restrict__ boxes,
                                              float* __restrict__ sc,
                                              float* __restrict__ rois,
                                              float* __restrict__ roiidx) {
    const int n   = blockIdx.x;
    const int tid = threadIdx.x;
    const float* bx = boxes + (size_t)n * 36864 * 4;
    float* s  = sc + (size_t)n * 36864;
    float* ro = rois + (size_t)n * 4000;
    float* ri = roiidx + (size_t)n * 1000;
    __shared__ float rv[32];
    __shared__ int   rix[32];
    __shared__ float bc[5];
    __shared__ int   sstop;
    int istop = 1000;

    for (int it = 0; it < 1000; ++it) {
        // argmax with first-index tie-break (matches jnp.argmax)
        float bv = NEGINF; int bi = 0x7fffffff;
        for (int step = 0; step < 36; ++step) {
            int k = tid + (step << 10);
            float v = s[k];
            if (v > bv || (v == bv && k < bi)) { bv = v; bi = k; }
        }
        for (int off = 16; off > 0; off >>= 1) {
            float ov = __shfl_down(bv, off, 32);
            int   oi = __shfl_down(bi, off, 32);
            if (ov > bv || (ov == bv && oi < bi)) { bv = ov; bi = oi; }
        }
        if ((tid & 31) == 0) { rv[tid >> 5] = bv; rix[tid >> 5] = bi; }
        __syncthreads();
        if (tid < 32) {
            bv = rv[tid]; bi = rix[tid];
            for (int off = 16; off > 0; off >>= 1) {
                float ov = __shfl_down(bv, off, 32);
                int   oi = __shfl_down(bi, off, 32);
                if (ov > bv || (ov == bv && oi < bi)) { bv = ov; bi = oi; }
            }
            if (tid == 0) {
                bool ok = bv > NEGINF;
                sstop = ok ? 0 : 1;
                if (ok) {
                    float b0 = bx[(long)bi * 4 + 0], b1 = bx[(long)bi * 4 + 1];
                    float b2 = bx[(long)bi * 4 + 2], b3 = bx[(long)bi * 4 + 3];
                    bc[0] = b0; bc[1] = b1; bc[2] = b2; bc[3] = b3;
                    bc[4] = (b2 - b0 + 1.f) * (b3 - b1 + 1.f);
                    ro[it * 4 + 0] = b0; ro[it * 4 + 1] = b1;
                    ro[it * 4 + 2] = b2; ro[it * 4 + 3] = b3;
                    ri[it] = (float)n;
                }
            }
        }
        __syncthreads();
        if (sstop) { istop = it; break; }
        const float jx1 = bc[0], jy1 = bc[1], jx2 = bc[2], jy2 = bc[3], ja = bc[4];
        for (int step = 0; step < 36; ++step) {
            int k = tid + (step << 10);
            float v = s[k];
            if (v > NEGINF) {
                float kx1 = bx[(long)k * 4 + 0], ky1 = bx[(long)k * 4 + 1];
                float kx2 = bx[(long)k * 4 + 2], ky2 = bx[(long)k * 4 + 3];
                float xx1 = fmaxf(jx1, kx1), yy1 = fmaxf(jy1, ky1);
                float xx2 = fminf(jx2, kx2), yy2 = fminf(jy2, ky2);
                float inter = fmaxf(0.f, xx2 - xx1 + 1.f) * fmaxf(0.f, yy2 - yy1 + 1.f);
                float ka = (kx2 - kx1 + 1.f) * (ky2 - ky1 + 1.f);
                float ovr = inter / (ja + ka - inter);
                if (ovr > 0.5f) s[k] = NEGINF;
            }
        }
        __threadfence();
        __syncthreads();
    }
    // fill unselected rows: rois=0, roi_index=-1
    for (int r = istop + tid; r < 1000; r += 1024) {
        ro[r * 4 + 0] = 0.f; ro[r * 4 + 1] = 0.f; ro[r * 4 + 2] = 0.f; ro[r * 4 + 3] = 0.f;
        ri[r] = -1.f;
    }
}

extern "C" void kernel_launch(void* const* d_in, const int* in_sizes, int n_in,
                              void* d_out, int out_size, void* d_ws, size_t ws_size,
                              hipStream_t stream) {
    (void)in_sizes; (void)n_in; (void)out_size; (void)ws_size;
    const float* x   = (const float*)d_in[0];
    const float* w1  = (const float*)d_in[1];
    const float* b1  = (const float*)d_in[2];
    const float* wsc = (const float*)d_in[3];
    const float* bs  = (const float*)d_in[4];
    const float* wlc = (const float*)d_in[5];
    const float* bl  = (const float*)d_in[6];
    const int*   imh = (const int*)d_in[7];
    const int*   imw = (const int*)d_in[8];

    char* base = (char*)d_ws;
    unsigned short* xpad   = (unsigned short*)(base);                 // 35,684,352 B
    v4u*            wp1    = (v4u*)(base + 35684352);                 //  4,718,592 B
    v4u*            wsp    = (v4u*)(base + 40402944);                 //     32,768 B
    v4u*            wlp    = (v4u*)(base + 40435712);                 //     49,152 B
    unsigned short* feat   = (unsigned short*)(base + 40484864);      // 33,554,432 B
    float*          boxes  = (float*)(base + 74039296);               //  4,718,592 B
    float*          scinit = (float*)(base + 78757888);               //  1,179,648 B

    float* out        = (float*)d_out;
    float* out_locs   = out;                 // 8*36864*4
    float* out_scores = out + 1179648;       // 8*36864*2
    float* out_rois   = out + 1769472;       // 8*1000*4
    float* out_ridx   = out + 1801472;       // 8*1000
    float* out_anchor = out + 1809472;       // 36864*4

    k_pad_x<<<69696, 256, 0, stream>>>(x, xpad);
    k_pack_w1<<<4608, 256, 0, stream>>>(w1, (unsigned*)wp1);
    k_pack_head<<<32, 256, 0, stream>>>(wsc, (unsigned*)wsp, 18, 2);
    k_pack_head<<<48, 256, 0, stream>>>(wlc, (unsigned*)wlp, 36, 3);
    k_conv1<<<dim3(32, 8, 8), 256, 0, stream>>>(xpad, wp1, b1, feat);
    k_head<<<dim3(256, 8), 256, 0, stream>>>(feat, wsp, wlp, bs, bl, imh, imw,
                                             out_locs, out_scores, out_anchor, boxes, scinit);
    k_nms<<<8, 1024, 0, stream>>>(boxes, scinit, out_rois, out_ridx);
}
